// GNNModel_833223656474
// MI455X (gfx1250) — matrix operand order repaired and checked by reading, not compiled, not run on hardware
//
#include <hip/hip_runtime.h>
#include <hip/hip_bf16.h>

// ---------------------------------------------------------------------------
// GCN (2-layer) for MI455X / gfx1250, wave32.
//   layer: t = A @ W (bf16 WMMA, f32 acc);  agg = bias; agg[dst] += norm * t[src]
// GEMMs (~5 GFLOP total) run on v_wmma_f32_16x16x32_bf16 with W pre-swizzled
// into LDS in fragment order (2x ds_load_b128 per B fragment) and A loaded as
// float4 -> v_cvt_pk_bf16_f32. The dominant message phase (2x ~870MB of
// random gathers + fp32 atomics) stays resident in the 192MB L2.
// ---------------------------------------------------------------------------

typedef __attribute__((ext_vector_type(16))) __bf16 v16bf;
typedef __attribute__((ext_vector_type(8)))  float  v8f;

#define F_IN 128
#define HID  128
#define F_OUT 64

// --------------------------- degree / normalization -------------------------

__global__ __launch_bounds__(256)
void degree_kernel(const int* __restrict__ dsts, float* __restrict__ deg, int E) {
    int i = blockIdx.x * blockDim.x + threadIdx.x;
    if (i < E)
        __hip_atomic_fetch_add(&deg[dsts[i]], 1.0f,
                               __ATOMIC_RELAXED, __HIP_MEMORY_SCOPE_AGENT);
}

__global__ __launch_bounds__(256)
void dinv_kernel(float* __restrict__ deg, int n) {
    int i = blockIdx.x * blockDim.x + threadIdx.x;
    if (i < n) {
        float d = deg[i] + 1.0f;          // +1 self-loop; always > 0
        deg[i] = rsqrtf(d);               // in-place: deg becomes dinv
    }
}

// ------------------------ bias broadcast initialization ---------------------

template<int F>
__global__ __launch_bounds__(256)
void fill_bias_kernel(float* __restrict__ out, const float* __restrict__ b, int n) {
    int i = blockIdx.x * blockDim.x + threadIdx.x;
    if (i < n * F) out[i] = b[i & (F - 1)];
}

// ------------------------------- WMMA GEMM ----------------------------------
// One wave computes one 16x16 output tile; block covers a 16-row stripe with
// NC/16 waves (one per column tile). K = 128 -> 4 WMMA steps of 32.
// W is staged once per block into LDS *already in B-fragment layout*
// (blds[ct][ks][lane][16 bf16]) so each lane reads its fragment as 32
// contiguous bytes (2x ds_load_b128), no repacking.

__device__ __forceinline__ float4 relu4(float4 v) {
    v.x = fmaxf(v.x, 0.0f); v.y = fmaxf(v.y, 0.0f);
    v.z = fmaxf(v.z, 0.0f); v.w = fmaxf(v.w, 0.0f);
    return v;
}

template<int K, int NC, bool RELU_A>
__global__ __launch_bounds__(32 * (NC / 16))
void gemm_wmma_bf16(const float* __restrict__ A, const float* __restrict__ W,
                    float* __restrict__ Out, int nRows) {
    constexpr int KSTEPS = K / 32;
    constexpr int CTILES = NC / 16;
    __shared__ __bf16 blds[CTILES * KSTEPS * 32 * 16];   // == K*NC elements

    // Cooperative fill, swizzled to fragment order:
    //   j = ((ct*KSTEPS + ks)*32 + L)*16 + i
    //   element = W[(ks*32 + (L>=16 ? 16:0) + i) * NC + ct*16 + (L&15)]
    for (int j = threadIdx.x; j < K * NC; j += blockDim.x) {
        int i    = j & 15;
        int L    = (j >> 4) & 31;
        int rest = j >> 9;                 // ct*KSTEPS + ks
        int ks   = rest & (KSTEPS - 1);
        int ct   = rest / KSTEPS;
        int k    = ks * 32 + ((L >= 16) ? 16 : 0) + i;
        int n    = ct * 16 + (L & 15);
        blds[j] = (__bf16)W[k * NC + n];
    }
    __syncthreads();

    const int wave   = threadIdx.x >> 5;       // column tile within stripe
    const int lane   = threadIdx.x & 31;
    const int laneLo = lane & 15;
    const bool hi    = lane >= 16;

    const int row0 = blockIdx.x * 16;
    int arow = row0 + laneLo;
    if (arow >= nRows) arow = nRows - 1;       // clamp: loads unconditional,
                                               // stores guarded below
    const float* __restrict__ ap = A + (size_t)arow * K;
    // A fragment (contiguous k order, matching B in LDS): lanes 0-15 hold K {0..15},
    // lanes 16-31 hold K {16..31}; element i in-VGPR-order.
    const int kA = hi ? 16 : 0;

    v8f acc = {};
#pragma unroll
    for (int ks = 0; ks < KSTEPS; ++ks) {
        const int kbase = ks * 32;
        float4 f0 = *(const float4*)(ap + kbase + kA);
        float4 f1 = *(const float4*)(ap + kbase + kA + 4);
        float4 f2 = *(const float4*)(ap + kbase + kA + 8);
        float4 f3 = *(const float4*)(ap + kbase + kA + 12);
        if (RELU_A) { f0 = relu4(f0); f1 = relu4(f1); f2 = relu4(f2); f3 = relu4(f3); }

        v16bf a;
        a[0]  = (__bf16)f0.x; a[1]  = (__bf16)f0.y; a[2]  = (__bf16)f0.z; a[3]  = (__bf16)f0.w;
        a[4]  = (__bf16)f1.x; a[5]  = (__bf16)f1.y; a[6]  = (__bf16)f1.z; a[7]  = (__bf16)f1.w;
        a[8]  = (__bf16)f2.x; a[9]  = (__bf16)f2.y; a[10] = (__bf16)f2.z; a[11] = (__bf16)f2.w;
        a[12] = (__bf16)f3.x; a[13] = (__bf16)f3.y; a[14] = (__bf16)f3.z; a[15] = (__bf16)f3.w;

        const v16bf b = *(const v16bf*)&blds[(((wave * KSTEPS) + ks) * 32 + lane) * 16];

        acc = __builtin_amdgcn_wmma_f32_16x16x32_bf16(
            /*neg_a=*/false, a, /*neg_b=*/false, b,
            /*c_mod=*/(short)0, acc, /*reuse_a=*/false, /*reuse_b=*/false);
    }

    // C/D layout: VGPR r -> M = r (lanes 0-15) or r+8 (lanes 16-31), N = laneLo
    const int mofs = hi ? 8 : 0;
    float* __restrict__ ocol = Out + (size_t)row0 * NC + wave * 16 + laneLo;
#pragma unroll
    for (int r = 0; r < 8; ++r) {
        int m = row0 + r + mofs;
        if (m < nRows) ocol[(size_t)(r + mofs) * NC] = acc[r];
    }
}

// ------------------------------ message scatter -----------------------------
// One wave per message (edge or self-loop). Lane handles F/32 consecutive
// features: coalesced gather from t[src] (L2-resident), f32 atomics to agg[dst].

template<int F>
__global__ __launch_bounds__(256)
void scatter_kernel(const int* __restrict__ srcs, const int* __restrict__ dsts,
                    const float* __restrict__ dinv, const float* __restrict__ t,
                    float* __restrict__ agg, int E, int nNodes) {
    const int wid  = blockIdx.x * (blockDim.x >> 5) + (threadIdx.x >> 5);
    const int lane = threadIdx.x & 31;
    const int nMsg = E + nNodes;
    if (wid >= nMsg) return;

    int s, d;
    if (wid < E) { s = srcs[wid]; d = dsts[wid]; }
    else         { s = wid - E;   d = s; }       // self-loop

    const float norm = dinv[s] * dinv[d];

    constexpr int FP = F / 32;                   // floats per lane (4 or 2)
    const float* __restrict__ trow = t   + (size_t)s * F + lane * FP;
    float*       __restrict__ arow = agg + (size_t)d * F + lane * FP;
#pragma unroll
    for (int i = 0; i < FP; ++i)
        __hip_atomic_fetch_add(&arow[i], trow[i] * norm,
                               __ATOMIC_RELAXED, __HIP_MEMORY_SCOPE_AGENT);
}

// ------------------------------- launcher -----------------------------------

extern "C" void kernel_launch(void* const* d_in, const int* in_sizes, int n_in,
                              void* d_out, int out_size, void* d_ws, size_t ws_size,
                              hipStream_t stream) {
    const float* x  = (const float*)d_in[0];
    const int*   ei = (const int*)  d_in[1];   // [2,E] flat: row0=src, row1=dst
    const float* W1 = (const float*)d_in[2];
    const float* b1 = (const float*)d_in[3];
    const float* W2 = (const float*)d_in[4];
    const float* b2 = (const float*)d_in[5];
    float* out = (float*)d_out;

    const int N = in_sizes[0] / F_IN;          // 100000
    const int E = in_sizes[1] / 2;             // 1600000
    const int* srcs = ei;
    const int* dsts = ei + E;

    // workspace: dinv[N] | t[N*128] (t1, reused for t2) | agg1[N*128]
    float* dinv = (float*)d_ws;
    float* t    = dinv + N;
    float* agg1 = t + (size_t)N * HID;

    const int nMsg = E + N;
    const int msgBlocks = (nMsg + 7) / 8;      // 8 waves / 256-thread block

    // degree -> dinv (in place)
    hipMemsetAsync(dinv, 0, (size_t)N * sizeof(float), stream);
    degree_kernel<<<(E + 255) / 256, 256, 0, stream>>>(dsts, dinv, E);
    dinv_kernel<<<(N + 255) / 256, 256, 0, stream>>>(dinv, N);

    // ----- layer 1: t1 = x @ W1 ; agg1 = b1 + scatter(norm * t1); relu folded into GEMM2
    fill_bias_kernel<HID><<<((size_t)N * HID + 255) / 256, 256, 0, stream>>>(agg1, b1, N);
    gemm_wmma_bf16<F_IN, HID, false>
        <<<(N + 15) / 16, 32 * (HID / 16), 0, stream>>>(x, W1, t, N);
    scatter_kernel<HID><<<msgBlocks, 256, 0, stream>>>(srcs, dsts, dinv, t, agg1, E, N);

    // ----- layer 2: t2 = relu(agg1) @ W2 ; out = b2 + scatter(norm * t2)
    gemm_wmma_bf16<HID, F_OUT, true>
        <<<(N + 15) / 16, 32 * (F_OUT / 16), 0, stream>>>(agg1, W2, t, N);
    fill_bias_kernel<F_OUT><<<((size_t)N * F_OUT + 255) / 256, 256, 0, stream>>>(out, b2, N);
    scatter_kernel<F_OUT><<<msgBlocks, 256, 0, stream>>>(srcs, dsts, dinv, t, out, E, N);
}
